// Attention_38087769981410
// MI455X (gfx1250) — matrix operand order repaired and checked by reading, not compiled, not run on hardware
//
#include <hip/hip_runtime.h>
#include <hip/hip_bf16.h>

typedef __bf16 bf16;
typedef __attribute__((ext_vector_type(16))) __bf16 v16bf;
typedef __attribute__((ext_vector_type(8)))  __bf16 v8bf;
typedef __attribute__((ext_vector_type(4)))  __bf16 v4bf;
typedef __attribute__((ext_vector_type(8)))  float  v8f;
typedef __attribute__((ext_vector_type(4)))  float  v4f;
typedef __attribute__((ext_vector_type(4)))  int    v4i;

#define HIDDEN 768
#define NHEADS 12
#define HDIM   64
#define BATCH  4
#define SEQ    2048

// CDNA5 async global->LDS path (guarded: falls back to regular loads if the
// toolchain doesn't declare the builtins).
#if defined(__gfx1250__) && __has_builtin(__builtin_amdgcn_global_load_async_to_lds_b128) && __has_builtin(__builtin_amdgcn_s_wait_asynccnt)
#define USE_ASYNC_LDS 1
#define GLOBAL_AS __attribute__((address_space(1)))
#define LDS_AS    __attribute__((address_space(3)))
#else
#define USE_ASYNC_LDS 0
#endif

static __device__ __forceinline__ v16bf cat16(v8bf a, v8bf b) {
  return __builtin_shufflevector(a, b, 0,1,2,3,4,5,6,7,8,9,10,11,12,13,14,15);
}
static __device__ __forceinline__ v8f wmma_bf16(v16bf a, v16bf b, v8f c) {
  return __builtin_amdgcn_wmma_f32_16x16x32_bf16(false, a, false, b, (short)0, c, false, false);
}

// ------------------------------------------------------------ convert + transpose
// x[b][c][s] f32  ->  Xt[b][s][c] bf16   (LDS-tiled 64x64 transpose)
__global__ __launch_bounds__(256)
void k_cvt_transpose(const float* __restrict__ x, bf16* __restrict__ Xt) {
  __shared__ __align__(16) bf16 T[64][65];     // padded to dodge bank conflicts
  const int t  = threadIdx.x;
  const int b  = blockIdx.z;
  const int c0 = blockIdx.y * 64;
  const int s0 = blockIdx.x * 64;
  #pragma unroll
  for (int i = 0; i < 4; ++i) {
    const int e4 = t + i * 256;                // 1024 float4 per tile
    const int c  = e4 >> 4;                    // 16 float4 per 64-wide row
    const int s4 = (e4 & 15) << 2;
    v4f f = *(const v4f*)(x + ((size_t)b * HIDDEN + c0 + c) * SEQ + s0 + s4);
    T[c][s4+0] = (bf16)f[0]; T[c][s4+1] = (bf16)f[1];
    T[c][s4+2] = (bf16)f[2]; T[c][s4+3] = (bf16)f[3];
  }
  __syncthreads();
  #pragma unroll
  for (int i = 0; i < 4; ++i) {
    const int e4 = t + i * 256;
    const int s  = e4 >> 4;
    const int c4 = (e4 & 15) << 2;
    v4bf h; h[0]=T[c4+0][s]; h[1]=T[c4+1][s]; h[2]=T[c4+2][s]; h[3]=T[c4+3][s];
    *(v4bf*)(Xt + ((size_t)b * SEQ + s0 + s) * HIDDEN + c0 + c4) = h;
  }
}

// ------------------------------------------------------------ GEMM (bf16 WMMA)
// Y[o,s] = sum_c W[o,c] * X[b,s,c] + bias[o]
// MODE 0: f32  [b][o][s]          (final output projection -> d_out)
// MODE 1: bf16 [b][o][s]          (V)
// MODE 2: bf16 [b][h][s][d]       (Q, K transposed for attention)
template<int MODE>
__global__ __launch_bounds__(128)
void k_gemm(const float* __restrict__ W, const float* __restrict__ bias,
            const bf16* __restrict__ Xt, void* __restrict__ Y) {
  __shared__ __align__(16) bf16 As[64][64];    // A tile [m][k], 8KB
  __shared__ __align__(16) bf16 Bs[64][64];    // B tile [n][k], 8KB
  const int t    = threadIdx.x;
  const int wave = t >> 5, lane = t & 31;
  const int hi   = (lane >> 4) & 1, l16 = lane & 15;
  const int m_blk = blockIdx.x * 64;
  const int n_blk = blockIdx.y * 64;
  const int b     = blockIdx.z;

  v8f acc[4] = {};

  for (int k0 = 0; k0 < HIDDEN; k0 += 64) {
    // ---- A fill: 64x64 f32 -> bf16 (1024 float4; 8 per thread)
    #pragma unroll
    for (int i = 0; i < 8; ++i) {
      const int e4  = t + i * 128;
      const int row = e4 >> 4;                 // 16 float4 per 64-wide row
      const int c4  = (e4 & 15) << 2;
      v4f f = *(const v4f*)(W + (size_t)(m_blk + row) * HIDDEN + k0 + c4);
      v4bf h; h[0]=(bf16)f[0]; h[1]=(bf16)f[1]; h[2]=(bf16)f[2]; h[3]=(bf16)f[3];
      *(v4bf*)(&As[row][c4]) = h;
    }
    // ---- B fill: straight row copy Xt[b][n_blk+row][k0..k0+64) -> Bs[row][*]
#if USE_ASYNC_LDS
    #pragma unroll
    for (int i = 0; i < 4; ++i) {
      const int ch  = t + i * 128;             // 512 16-byte chunks
      const int row = ch >> 3;                 // 8 chunks per 128B row
      const int off = (ch & 7) << 3;
      const bf16* g = Xt + ((size_t)b * SEQ + n_blk + row) * HIDDEN + k0 + off;
      __builtin_amdgcn_global_load_async_to_lds_b128(
          (GLOBAL_AS v4i*)g, (LDS_AS v4i*)(&Bs[row][off]), 0, 0);
    }
    __builtin_amdgcn_s_wait_asynccnt(0);
#else
    #pragma unroll
    for (int i = 0; i < 4; ++i) {
      const int ch  = t + i * 128;
      const int row = ch >> 3;
      const int off = (ch & 7) << 3;
      v8bf h = *(const v8bf*)(Xt + ((size_t)b * SEQ + n_blk + row) * HIDDEN + k0 + off);
      *(v8bf*)(&Bs[row][off]) = h;
    }
#endif
    __syncthreads();

    const int arow = wave * 16 + l16;
    #pragma unroll
    for (int kc = 0; kc < 2; ++kc) {
      // A fragment: row m; K pattern {0-7,16-23}/{8-15,24-31} within 32-chunk
      v16bf af = cat16(*(const v8bf*)(&As[arow][kc*32 + hi*8]),
                       *(const v8bf*)(&As[arow][kc*32 + 16 + hi*8]));
      #pragma unroll
      for (int j = 0; j < 4; ++j) {
        // B fragment: col n; K pattern {0-7,16-23}/{8-15,24-31} within 32-chunk
        const int bcol = j * 16 + l16;
        v16bf bfr = cat16(*(const v8bf*)(&Bs[bcol][kc*32 + hi*8]),
                          *(const v8bf*)(&Bs[bcol][kc*32 + 16 + hi*8]));
        acc[j] = wmma_bf16(af, bfr, acc[j]);
      }
    }
    __syncthreads();
  }

  // epilogue (fully unrolled: no indexed-VGPR access)
  #pragma unroll
  for (int r = 0; r < 8; ++r) {
    const int o = m_blk + wave * 16 + r + hi * 8;
    const float bv = bias[o];
    #pragma unroll
    for (int j = 0; j < 4; ++j) {
      const int n = n_blk + j * 16 + l16;
      const float val = acc[j][r] + bv;
      if (MODE == 0) {
        ((float*)Y)[((size_t)b * HIDDEN + o) * SEQ + n] = val;
      } else if (MODE == 1) {
        ((bf16*)Y)[((size_t)b * HIDDEN + o) * SEQ + n] = (bf16)val;
      } else {
        const int h = o >> 6, d = o & 63;
        ((bf16*)Y)[(((size_t)b * NHEADS + h) * SEQ + n) * HDIM + d] = (bf16)val;
      }
    }
  }
}

// ------------------------------------------------------------ attention
// One wave per (b, h, 16-query tile); flash-style online softmax.
// scores[k,q] = (1/8)*sum_d K[d,k]*Q[d,q] + mask[b,k,q]; attn[d,q]=sum_k w*V
__global__ __launch_bounds__(128)
void k_attn(const bf16* __restrict__ Qt, const bf16* __restrict__ Kt,
            const bf16* __restrict__ V,  const float* __restrict__ mask,
            bf16* __restrict__ O) {
  const int t    = threadIdx.x;
  const int lane = t & 31, wave = t >> 5;
  const int hi   = (lane >> 4) & 1, l16 = lane & 15;
  const int gw   = blockIdx.x * 4 + wave;
  const int qtile = gw & 127;                  // SEQ/16 tiles
  const int h     = (gw >> 7) % NHEADS;
  const int b     = gw / (128 * NHEADS);
  const int q0    = qtile * 16;

  const size_t bh = (size_t)b * NHEADS + h;
  const bf16* qb = Qt + (bh * SEQ + q0 + l16) * HDIM;
  const v16bf qf0 = cat16(*(const v8bf*)(qb + hi*8),      *(const v8bf*)(qb + 16 + hi*8));
  const v16bf qf1 = cat16(*(const v8bf*)(qb + 32 + hi*8), *(const v8bf*)(qb + 48 + hi*8));

  const bf16*  vbase = V + ((size_t)b * HIDDEN + h * HDIM) * SEQ;
  const float* mbase = mask + (size_t)b * SEQ * SEQ;

  v8f acc[4] = {};
  float mq = -INFINITY, lq = 0.f;
  const float scale = 0.125f;                  // HEAD_DIM^-0.5

  for (int k0 = 0; k0 < SEQ; k0 += 32) {
    // prefetch next chunk's 32 mask rows (lane -> one row each)
    if (k0 + 32 < SEQ)
      __builtin_prefetch(mbase + (size_t)(k0 + 32 + lane) * SEQ + q0, 0, 1);

    // scores: two 16(k)x16(q) tiles, d=64 reduced by two chained WMMAs each
    v8f s0, s1;
    {
      const bf16* kb = Kt + (bh * SEQ + k0 + l16) * HDIM;
      v16bf ka0 = cat16(*(const v8bf*)(kb + hi*8),      *(const v8bf*)(kb + 16 + hi*8));
      v16bf ka1 = cat16(*(const v8bf*)(kb + 32 + hi*8), *(const v8bf*)(kb + 48 + hi*8));
      v8f z = {};
      z = wmma_bf16(ka0, qf0, z);
      s0 = wmma_bf16(ka1, qf1, z);
    }
    {
      const bf16* kb = Kt + (bh * SEQ + k0 + 16 + l16) * HDIM;
      v16bf ka0 = cat16(*(const v8bf*)(kb + hi*8),      *(const v8bf*)(kb + 16 + hi*8));
      v16bf ka1 = cat16(*(const v8bf*)(kb + 32 + hi*8), *(const v8bf*)(kb + 48 + hi*8));
      v8f z = {};
      z = wmma_bf16(ka0, qf0, z);
      s1 = wmma_bf16(ka1, qf1, z);
    }

    // scale + mask; online-softmax stats (lanes l16 / l16+16 share column q)
    float p0[8], p1[8];
    float cmax = -INFINITY;
    #pragma unroll
    for (int r = 0; r < 8; ++r) {
      const int kr = k0 + r + hi * 8;
      const float m0 = mbase[(size_t)kr        * SEQ + q0 + l16];
      const float m1 = mbase[(size_t)(kr + 16) * SEQ + q0 + l16];
      p0[r] = s0[r] * scale + m0;
      p1[r] = s1[r] * scale + m1;
      cmax = fmaxf(cmax, fmaxf(p0[r], p1[r]));
    }
    cmax = fmaxf(cmax, __shfl_xor(cmax, 16, 32));
    const float newm  = fmaxf(mq, cmax);
    const float alpha = __expf(mq - newm);
    mq = newm;
    float psum = 0.f;
    #pragma unroll
    for (int r = 0; r < 8; ++r) {
      p0[r] = __expf(p0[r] - newm);
      p1[r] = __expf(p1[r] - newm);
      psum += p0[r] + p1[r];
    }
    psum += __shfl_xor(psum, 16, 32);
    lq = lq * alpha + psum;
    #pragma unroll
    for (int j = 0; j < 4; ++j)
      #pragma unroll
      for (int r = 0; r < 8; ++r) acc[j][r] *= alpha;

    // repack exp'd scores (D layout) into B-fragment layout (K blocks of 16/half)
    v16bf w;
    #pragma unroll
    for (int r = 0; r < 8; ++r) {
      const float x0 = __shfl_xor(p0[r], 16, 32);
      const float x1 = __shfl_xor(p1[r], 16, 32);
      w[r]     = (bf16)(hi ? x1    : p0[r]);   // K = k0 + (hi?16+r : r)
      w[r + 8] = (bf16)(hi ? p1[r] : x0);      // K = k0 + (hi?24+r : 8+r)
    }
    // PV: attn[d,q] += V[d,k] * w[k,q]
    #pragma unroll
    for (int j = 0; j < 4; ++j) {
      const bf16* vb = vbase + (size_t)(j * 16 + l16) * SEQ + k0;
      v16bf va = cat16(*(const v8bf*)(vb + hi*16), *(const v8bf*)(vb + hi*16 + 8));
      acc[j] = wmma_bf16(va, w, acc[j]);
    }
  }

  // normalize + store to At[b][s][c]: per lane 4 contiguous 16B stores
  const float inv = 1.0f / lq;
  bf16* obase = O + ((size_t)b * SEQ + q0 + l16) * HIDDEN + h * HDIM;
  #pragma unroll
  for (int j = 0; j < 4; ++j) {
    v8bf o8;
    #pragma unroll
    for (int r = 0; r < 8; ++r) o8[r] = (bf16)(acc[j][r] * inv);
    *(v8bf*)(obase + j * 16 + hi * 8) = o8;
  }
}

// ------------------------------------------------------------ launch
extern "C" void kernel_launch(void* const* d_in, const int* in_sizes, int n_in,
                              void* d_out, int out_size, void* d_ws, size_t ws_size,
                              hipStream_t stream) {
  const float* x    = (const float*)d_in[0];
  const float* mask = (const float*)d_in[1];
  const float* Wq   = (const float*)d_in[2];
  const float* bq   = (const float*)d_in[3];
  const float* Wk   = (const float*)d_in[4];
  const float* bk   = (const float*)d_in[5];
  const float* Wv   = (const float*)d_in[6];
  const float* bv   = (const float*)d_in[7];
  const float* Wo   = (const float*)d_in[8];
  const float* bo   = (const float*)d_in[9];
  float* out = (float*)d_out;

  const size_t NELEM = (size_t)BATCH * HIDDEN * SEQ;   // 6,291,456
  bf16* Xt = (bf16*)d_ws;                              // x   [b][s][c] bf16
  bf16* Qt = Xt + NELEM;                               // Q   [b][h][s][d]
  bf16* Kt = Qt + NELEM;                               // K   [b][h][s][d]
  bf16* Vb = Kt + NELEM;                               // V   [b][c][s]
  bf16* At = Vb + NELEM;                               // attn [b][s][c]

  k_cvt_transpose<<<dim3(SEQ / 64, HIDDEN / 64, BATCH), 256, 0, stream>>>(x, Xt);

  dim3 gg(HIDDEN / 64, SEQ / 64, BATCH), bb(128);
  k_gemm<2><<<gg, bb, 0, stream>>>(Wq, bq, Xt, Qt);
  k_gemm<2><<<gg, bb, 0, stream>>>(Wk, bk, Xt, Kt);
  k_gemm<1><<<gg, bb, 0, stream>>>(Wv, bv, Xt, Vb);

  const int nwaves = BATCH * NHEADS * (SEQ / 16);      // 6144
  k_attn<<<dim3(nwaves / 4), dim3(128), 0, stream>>>(Qt, Kt, Vb, mask, At);

  k_gemm<0><<<gg, bb, 0, stream>>>(Wo, bo, At, out);
}
